// RNNScratch_60490319396979
// MI455X (gfx1250) — matrix operand order repaired and checked by reading, compiled, not run on hardware
//
#include <hip/hip_runtime.h>
#include <math.h>

// ---------------------------------------------------------------------------
// RNN scratch for MI455X (gfx1250, wave32, WMMA).
//   Phase 1: xw = inputs @ W_xh + b   (big GEMM, bf16 WMMA, f32 accum,
//            software-pipelined LDS staging of A)
//   Phase 2: h_t = tanh(xw_t + h_{t-1} @ W_hh)  (persistent cooperative scan,
//            W_hh slice resident in LDS, async-to-LDS copy when available)
// ---------------------------------------------------------------------------

typedef __attribute__((ext_vector_type(16))) __bf16        v16bf;
typedef __attribute__((ext_vector_type(8)))  float         v8f;
typedef __attribute__((ext_vector_type(8)))  unsigned int  v8u;
typedef __attribute__((ext_vector_type(4)))  unsigned int  v4u;
typedef __attribute__((ext_vector_type(4)))  int           v4i;

#define T_STEPS   512
#define BATCH     256
#define HID       1024
#define KSTEPS    32          // HID / 32 (bf16 WMMA K)
#define SCAN_WGS  32          // each owns a 32-wide slice of H

static __device__ __forceinline__ v16bf as_v16bf(v8u x) {
  return __builtin_bit_cast(v16bf, x);
}
static __device__ __forceinline__ unsigned pk_bf16(float a, float b) {
  unsigned ua = __builtin_bit_cast(unsigned, a);
  unsigned ub = __builtin_bit_cast(unsigned, b);
  return (ua >> 16) | (ub & 0xffff0000u);       // v_perm-style pack (trunc)
}
static __device__ __forceinline__ unsigned short bf16_bits(float a) {
  return (unsigned short)(__builtin_bit_cast(unsigned, a) >> 16);
}
static __device__ __forceinline__ v8f wmma_bf16(v16bf a, v16bf b, v8f c) {
  // (neg_a, A, neg_b, B, c_mod, C, reuse_a, reuse_b)
  return __builtin_amdgcn_wmma_f32_16x16x32_bf16(false, a, false, b,
                                                 (short)0, c, false, false);
}
static __device__ __forceinline__ float fast_tanh(float x) {
#if __has_builtin(__builtin_amdgcn_tanhf)
  return __builtin_amdgcn_tanhf(x);
#elif __has_builtin(__builtin_amdgcn_tanh_f32)
  return __builtin_amdgcn_tanh_f32(x);
#else
  return tanhf(x);
#endif
}

// ---------------------------------------------------------------------------
// Swizzle a [1024 x 1024] f32 weight matrix into bf16 WMMA B-fragment layout.
// Fragment = 32(K) x 16(N): lane = (n&15) + 16*((k>>3)&1), elem = (k&7) + 8*((k>>4)&1),
// fragment index = (k>>5)*64 + (n>>4), 1024 bytes per fragment.
// ---------------------------------------------------------------------------
__global__ __launch_bounds__(256) void swizzle_w(const float* __restrict__ W,
                                                 unsigned short* __restrict__ out) {
  int idx = blockIdx.x * 256 + threadIdx.x;     // 1M elements
  int k = idx >> 10;
  int n = idx & (HID - 1);
  unsigned short v = bf16_bits(W[idx]);
  unsigned lane = (unsigned)(n & 15) | (((unsigned)(k >> 3) & 1u) << 4);
  unsigned elem = (unsigned)(k & 7) | (((unsigned)(k >> 4) & 1u) << 3);
  unsigned frag = (unsigned)(k >> 5) * 64u + (unsigned)(n >> 4);
  out[frag * 512u + lane * 16u + elem] = v;     // ushort units (32B per lane)
}

// ---------------------------------------------------------------------------
// Phase 1: C[131072,1024] = A[131072,1024] @ Bfrag(+bias). Tile 256x128,
// 8 waves (4M x 2N), each wave 64x64 = 4x4 WMMA tiles. A staging is
// software-pipelined: k-step ks+1 global loads issue before ks's WMMAs.
// ---------------------------------------------------------------------------
__global__ __launch_bounds__(256) void gemm_xw(const float* __restrict__ A,
                                               const unsigned short* __restrict__ Bfrag,
                                               const float* __restrict__ bias,
                                               float* __restrict__ C) {
  __shared__ __align__(16) unsigned char ldsA[16 * 1024];   // 16 A-frags / kstep

  const int tid  = threadIdx.x;
  const int lane = tid & 31;
  const int wv   = tid >> 5;          // 0..7
  const int wm   = wv & 3;            // 4 wave rows
  const int wn   = wv >> 2;           // 2 wave cols
  const long row0 = (long)blockIdx.y * 256;
  const int  col0 = (int)blockIdx.x * 128;

  // A staging: thread -> (row srow+32i, k chunk of 4), coalesced 128B per row
  const int srow = tid >> 3;          // 0..31
  const int sk4  = (tid & 7) * 4;     // 0,4,...,28
  const unsigned slane = (unsigned)(srow & 15) | (((sk4 & 15) >= 8) ? 16u : 0u);
  const unsigned selem = (unsigned)(sk4 & 7) | (((unsigned)(sk4 >> 4) & 1u) << 3);
  const unsigned sbyte = slane * 32u + selem * 2u;

  v8f acc[4][4];
  const v8f vzero = {0.f, 0.f, 0.f, 0.f, 0.f, 0.f, 0.f, 0.f};
#pragma unroll
  for (int i = 0; i < 4; ++i)
#pragma unroll
    for (int j = 0; j < 4; ++j) acc[i][j] = vzero;

  // pipeline prologue: fetch k-step 0
  float4 fA[8];
#pragma unroll
  for (int i = 0; i < 8; ++i)
    fA[i] = *(const float4*)(A + (row0 + srow + i * 32) * HID + sk4);

  for (int ks = 0; ks < KSTEPS; ++ks) {
    __syncthreads();
    // stage prefetched A tile [256 x 32] f32 -> bf16 fragments in LDS
#pragma unroll
    for (int i = 0; i < 8; ++i) {
      int r = srow + i * 32;
      uint2 d;
      d.x = pk_bf16(fA[i].x, fA[i].y);
      d.y = pk_bf16(fA[i].z, fA[i].w);
      *(uint2*)(ldsA + (unsigned)(r >> 4) * 1024u + sbyte) = d;
    }
    __syncthreads();

    // issue next k-step's global loads now; their wait lands on the next
    // iteration's ds_store, overlapping with this iteration's WMMAs
    const int ksn = (ks + 1 < KSTEPS) ? ks + 1 : ks;
#pragma unroll
    for (int i = 0; i < 8; ++i)
      fA[i] = *(const float4*)(A + (row0 + srow + i * 32) * HID + ksn * 32 + sk4);

    v16bf afr[4], bfr[4];
#pragma unroll
    for (int i = 0; i < 4; ++i) {
      int mblk = wm * 4 + i;
      afr[i] = as_v16bf(*(const v8u*)(ldsA + (unsigned)mblk * 1024u + (unsigned)lane * 32u));
    }
#pragma unroll
    for (int j = 0; j < 4; ++j) {
      int nsub = (col0 >> 4) + wn * 4 + j;
      bfr[j] = as_v16bf(*(const v8u*)(Bfrag + ((unsigned)(ks * 64 + nsub) * 512u +
                                               (unsigned)lane * 16u)));
    }
#pragma unroll
    for (int i = 0; i < 4; ++i)
#pragma unroll
      for (int j = 0; j < 4; ++j) acc[i][j] = wmma_bf16(afr[i], bfr[j], acc[i][j]);
  }

  // epilogue: + bias, store f32 (C/D layout: N=lane&15, M=r+8*(lane>>4))
  const int cN = lane & 15;
  const int cH = lane >> 4;
#pragma unroll
  for (int j = 0; j < 4; ++j) {
    int col = col0 + (wn * 4 + j) * 16 + cN;
    float bv = bias[col];
#pragma unroll
    for (int i = 0; i < 4; ++i) {
      long rowb = row0 + (wm * 4 + i) * 16 + 8 * cH;
#pragma unroll
      for (int r = 0; r < 8; ++r)
        C[(rowb + r) * HID + col] = acc[i][j][r] + bv;
    }
  }
}

// ---------------------------------------------------------------------------
// Phase 2: persistent cooperative scan. 32 WGs x 256 threads; WG w owns
// columns [32w, 32w+32). W_hh slice (64KB of fragments) lives in LDS.
// h_{t-1} is consumed as pre-swizzled bf16 A-fragments from a ping-pong
// buffer; each WG writes the K-chunk ks==wg (its 32 columns) for all M.
// ---------------------------------------------------------------------------
__global__ __launch_bounds__(256) void rnn_scan(const unsigned short* __restrict__ WhhFrag,
                                                float* __restrict__ out,
                                                unsigned short* __restrict__ hfrag,
                                                unsigned int* __restrict__ bar) {
  __shared__ __align__(16) unsigned char ldsB[64 * 1024];   // [ks][2][1024]

  const int tid  = threadIdx.x;
  const int lane = tid & 31;
  const int wv   = tid >> 5;            // 0..7 -> msubs {2wv, 2wv+1}
  const int wg   = blockIdx.x;          // 0..31
  const int n0   = wg * 32;

  // one-time copy of this WG's W_hh fragment slice into LDS (4096 x 16B)
#if __has_builtin(__builtin_amdgcn_global_load_async_to_lds_b128)
  {
    for (int i = tid; i < 4096; i += 256) {
      int ks = i >> 7, rem = i & 127;
      int j = rem >> 6, c = rem & 63;
      v4i* gp = (v4i*)((const unsigned char*)WhhFrag +
                       (unsigned)(ks * 64 + wg * 2 + j) * 1024u + (unsigned)c * 16u);
      v4i* lp = (v4i*)(ldsB + (unsigned)ks * 2048u + (unsigned)j * 1024u +
                       (unsigned)c * 16u);
      __builtin_amdgcn_global_load_async_to_lds_b128(gp, lp, 0, 0);
    }
#if __has_builtin(__builtin_amdgcn_s_wait_asynccnt)
    __builtin_amdgcn_s_wait_asynccnt(0);
#else
    asm volatile("s_wait_asynccnt 0x0" ::: "memory");
#endif
  }
#else
  for (int i = tid; i < 4096; i += 256) {
    int ks = i >> 7, rem = i & 127;
    int j = rem >> 6, c = rem & 63;
    const v4u s = *(const v4u*)((const unsigned char*)WhhFrag +
                                (unsigned)(ks * 64 + wg * 2 + j) * 1024u + (unsigned)c * 16u);
    *(v4u*)(ldsB + (unsigned)ks * 2048u + (unsigned)j * 1024u + (unsigned)c * 16u) = s;
  }
#endif
  __syncthreads();

  const int cN = lane & 15;
  const int cH = lane >> 4;
  const v8f vzero = {0.f, 0.f, 0.f, 0.f, 0.f, 0.f, 0.f, 0.f};
  const unsigned HSZ = (unsigned)KSTEPS * 16u * 512u;   // ping-pong stride (ushorts)

  for (int t = 0; t < T_STEPS; ++t) {
    v8f acc[2][2];
#pragma unroll
    for (int i = 0; i < 2; ++i)
#pragma unroll
      for (int j = 0; j < 2; ++j) acc[i][j] = vzero;

    if (t > 0) {
      const unsigned short* hsrc = hfrag + (unsigned)((t - 1) & 1) * HSZ;
#pragma unroll 4
      for (int ks = 0; ks < KSTEPS; ++ks) {
        v16bf a0 = as_v16bf(*(const v8u*)(hsrc + ((unsigned)(ks * 16 + wv * 2 + 0) * 512u +
                                                  (unsigned)lane * 16u)));
        v16bf a1 = as_v16bf(*(const v8u*)(hsrc + ((unsigned)(ks * 16 + wv * 2 + 1) * 512u +
                                                  (unsigned)lane * 16u)));
        v16bf b0 = as_v16bf(*(const v8u*)(ldsB + (unsigned)ks * 2048u + (unsigned)lane * 32u));
        v16bf b1 = as_v16bf(*(const v8u*)(ldsB + (unsigned)ks * 2048u + 1024u + (unsigned)lane * 32u));
        acc[0][0] = wmma_bf16(a0, b0, acc[0][0]);
        acc[0][1] = wmma_bf16(a0, b1, acc[0][1]);
        acc[1][0] = wmma_bf16(a1, b0, acc[1][0]);
        acc[1][1] = wmma_bf16(a1, b1, acc[1][1]);
      }
    }

    // h = tanh(acc + xw); overwrite out[t] with h; emit next-step A fragments
    float* outT = out + (long)t * (BATCH * HID);
    unsigned short* hdst = hfrag + (unsigned)(t & 1) * HSZ;
#pragma unroll
    for (int i = 0; i < 2; ++i) {
      int ms = wv * 2 + i;
#pragma unroll
      for (int j = 0; j < 2; ++j) {
        int col = n0 + j * 16 + cN;
        int kl  = j * 16 + cN;                    // local K within this 32-chunk
        unsigned lanep = (((kl & 15) >= 8) ? 16u : 0u);
        unsigned elemp = (unsigned)(kl & 7) | (((unsigned)(kl >> 4) & 1u) << 3);
#pragma unroll
        for (int r = 0; r < 8; ++r) {
          int m = ms * 16 + r + 8 * cH;
          float x = outT[(long)m * HID + col];
          float h = fast_tanh(acc[i][j][r] + x);
          outT[(long)m * HID + col] = h;
          if (t == T_STEPS - 1) {
            out[(long)T_STEPS * BATCH * HID + (long)m * HID + col] = h;
          } else {
            unsigned lp = (unsigned)(m & 15) | lanep;
            hdst[(unsigned)(wg * 16 + (m >> 4)) * 512u + lp * 16u + elemp] = bf16_bits(h);
          }
        }
      }
    }

    // device-wide barrier (monotonic counter; zeroed by memset each launch)
    __threadfence();
    __syncthreads();
    if (tid == 0) {
      atomicAdd(bar, 1u);
      unsigned tgt = (unsigned)(t + 1) * SCAN_WGS;
      while (__atomic_load_n(bar, __ATOMIC_RELAXED) < tgt) {
        __builtin_amdgcn_s_sleep(1);
      }
    }
    __syncthreads();
    __threadfence();
  }
}

// ---------------------------------------------------------------------------
// ws layout: [0,256) barrier counter | [256, +2MB) W_xh frags |
//            [+2MB, +4MB) W_hh frags | [+4MB, +5MB) h ping-pong frags
// ---------------------------------------------------------------------------
extern "C" void kernel_launch(void* const* d_in, const int* in_sizes, int n_in,
                              void* d_out, int out_size, void* d_ws, size_t ws_size,
                              hipStream_t stream) {
  (void)in_sizes; (void)n_in; (void)out_size; (void)ws_size;
  const float* inputs = (const float*)d_in[0];   // [512,256,1024]
  const float* W_xh   = (const float*)d_in[1];   // [1024,1024]
  const float* W_hh   = (const float*)d_in[2];   // [1024,1024]
  const float* b_h    = (const float*)d_in[3];   // [1024]
  float* out = (float*)d_out;

  unsigned char* ws = (unsigned char*)d_ws;
  unsigned int*   cnt   = (unsigned int*)ws;
  unsigned short* wxh_f = (unsigned short*)(ws + 256);
  unsigned short* whh_f = (unsigned short*)(ws + 256 + (2u << 20));
  unsigned short* hpp   = (unsigned short*)(ws + 256 + (4u << 20));

  (void)hipMemsetAsync(cnt, 0, 256, stream);
  swizzle_w<<<dim3(4096), 256, 0, stream>>>(W_xh, wxh_f);
  swizzle_w<<<dim3(4096), 256, 0, stream>>>(W_hh, whh_f);
  gemm_xw<<<dim3(HID / 128, (T_STEPS * BATCH) / 256), 256, 0, stream>>>(inputs, wxh_f, b_h, out);
  rnn_scan<<<dim3(SCAN_WGS), 256, 0, stream>>>(whh_f, out, hpp, cnt);
}
